// Model_44590350467369
// MI455X (gfx1250) — matrix operand order repaired and checked by reading, not compiled, not run on hardware
//
#include <hip/hip_runtime.h>
#include <hip/hip_bf16.h>
#include <math.h>
#include <stdint.h>

// ---------------- model constants ----------------
#define Bn 64
#define Ln 512
#define Hn 300
#define H2 600
#define G4 1200
#define Nn (Bn * Ln)       // 32768
#define En 524288
#define Rn 4
#define STEPS 2
#define Cn 20
#define XP 304             // padded bf16 row length for async staging (608 B)

typedef __attribute__((ext_vector_type(16))) __bf16 v16bf;
typedef __attribute__((ext_vector_type(8)))  __bf16 v8bf;
typedef __attribute__((ext_vector_type(4)))  __bf16 v4bf;
typedef __attribute__((ext_vector_type(8)))  float  v8f;
typedef int i32x4_ __attribute__((__vector_size__(16)));

union U16bf { v16bf v; v8bf h[2]; };

#if __has_builtin(__builtin_amdgcn_global_load_async_to_lds_b128)
#define HAVE_ASYNC_LDS 1
#endif

__device__ __forceinline__ void wait_asynccnt0() {
#if __has_builtin(__builtin_amdgcn_s_wait_asynccnt)
  __builtin_amdgcn_s_wait_asynccnt(0);
#else
  asm volatile("s_wait_asynccnt 0x0" ::: "memory");
#endif
}

#define WMMA_BF16(A_, B_, C_) \
  __builtin_amdgcn_wmma_f32_16x16x32_bf16(false, (A_), false, (B_), (short)0, (C_), false, false)

__device__ __forceinline__ float sigmoidf_(float x) { return 1.0f / (1.0f + __expf(-x)); }

__device__ __forceinline__ v8bf pack8(float4 a, float4 b) {
  v8bf r;
  r[0] = (__bf16)a.x; r[1] = (__bf16)a.y; r[2] = (__bf16)a.z; r[3] = (__bf16)a.w;
  r[4] = (__bf16)b.x; r[5] = (__bf16)b.y; r[6] = (__bf16)b.z; r[7] = (__bf16)b.w;
  return r;
}

// =================================================================
// Generic bf16-WMMA GEMM:  D[z] = act( alpha*(A[z] @ B[z]) + bias + beta*D[z] )
// A: [M,K] fp32 (optional row gather), B: [K,N] fp32 (or [N,K] if btrans)
// block = 256 threads (8 waves), macro tile 64x64, each wave does 16x32.
// LDS tiles stored in WMMA fragment order: one contiguous 32-byte read/frag.
//   A(m,k) -> Aswz[m>>4][(m&15)+16*((k>>4)&1)][k&15]
//   B(k,n) -> Bswz[n>>4][(n&15)+16*((k>>4)&1)][k&15]
// =================================================================
__global__ void k_gemm(const float* __restrict__ A, const float* __restrict__ Bw,
                       const float* __restrict__ bias, float* __restrict__ D,
                       int M, int Nc, int K, int lda, int ldb, int ldd,
                       long long sA, long long sB, long long sD,
                       const int* __restrict__ gather,
                       int btrans, float alpha, float beta, int act)
{
  __shared__ alignas(32) __bf16 Aswz[4][32][16];
  __shared__ alignas(32) __bf16 Bswz[4][32][16];

  const int tid  = threadIdx.x;
  const int wave = tid >> 5;
  const int lane = tid & 31;
  const int m0 = blockIdx.y * 64;
  const int n0 = blockIdx.x * 64;
  const long long z = blockIdx.z;
  A  += z * sA;  Bw += z * sB;  D += z * sD;

  const int wm = wave & 3;   // m-tile of wave
  const int wn = wave >> 2;  // n-group (2 tiles)

  // ---- A writer coords (per thread fixed) ----
  const int ar  = tid >> 2;          // m row 0..63
  const int ac0 = (tid & 3) << 3;    // k offset {0,8,16,24}
  const float* Abase = nullptr;
  {
    int gm = m0 + ar;
    if (gm < M) {
      long long arow = gather ? (long long)gather[gm] : (long long)gm;
      Abase = A + arow * (long long)lda;
    }
  }
  __bf16* adst = &Aswz[ar >> 4][(ar & 15) + ((ac0 >> 4) << 4)][ac0 & 15];

  // ---- B writer coords ----
  const int btn  = tid >> 2;         // btrans: n row 0..63
  const int btk0 = (tid & 3) << 3;   // btrans: k offset
  __bf16* btdst  = &Bswz[btn >> 4][(btn & 15) + (((btk0 >> 4) & 1) << 4)][btk0 & 15];
  const int nbk  = tid >> 3;         // normal: k row 0..31
  const int nbc0 = (tid & 7) << 3;   // normal: n base
  __bf16* nbdst  = &Bswz[nbc0 >> 4][(nbc0 & 15) + (((nbk >> 4) & 1) << 4)][nbk & 15];

  v8f acc0 = {0.f,0.f,0.f,0.f,0.f,0.f,0.f,0.f};
  v8f acc1 = {0.f,0.f,0.f,0.f,0.f,0.f,0.f,0.f};

  const int nK = (K + 31) >> 5;
  for (int kc = 0; kc < nK; ++kc) {
    const int k0 = kc << 5;
    // ---- stage A tile ----
    if (Abase && (k0 + ac0 + 8 <= K)) {
      const float4* p = (const float4*)(Abase + k0 + ac0);
      *(v8bf*)adst = pack8(p[0], p[1]);
    } else {
      v8bf r;
#pragma unroll
      for (int c = 0; c < 8; ++c) {
        int k = k0 + ac0 + c;
        float v = (Abase && k < K) ? Abase[k] : 0.0f;
        r[c] = (__bf16)v;
      }
      *(v8bf*)adst = r;
    }
    // ---- stage B tile ----
    if (btrans) {
      int gn = n0 + btn;
      const float* Bb = Bw + (long long)gn * ldb;
      if (gn < Nc && (k0 + btk0 + 8 <= K)) {
        const float4* p = (const float4*)(Bb + k0 + btk0);
        *(v8bf*)btdst = pack8(p[0], p[1]);
      } else {
        v8bf r;
#pragma unroll
        for (int c = 0; c < 8; ++c) {
          int k = k0 + btk0 + c;
          float v = (gn < Nc && k < K) ? Bb[k] : 0.0f;
          r[c] = (__bf16)v;
        }
        *(v8bf*)btdst = r;
      }
      if (gn < Nc && k0 + 32 + btk0 < K)
        __builtin_prefetch(Bb + k0 + 32 + btk0, 0, 3);
    } else {
      int gk = k0 + nbk;
      const float* Bb = Bw + (long long)gk * ldb + n0 + nbc0;
      if (gk < K && (n0 + nbc0 + 8 <= Nc)) {
        const float4* p = (const float4*)Bb;
        float4 f0 = p[0], f1 = p[1];
        nbdst[0*16] = (__bf16)f0.x; nbdst[1*16] = (__bf16)f0.y;
        nbdst[2*16] = (__bf16)f0.z; nbdst[3*16] = (__bf16)f0.w;
        nbdst[4*16] = (__bf16)f1.x; nbdst[5*16] = (__bf16)f1.y;
        nbdst[6*16] = (__bf16)f1.z; nbdst[7*16] = (__bf16)f1.w;
      } else {
#pragma unroll
        for (int c = 0; c < 8; ++c) {
          int gn = n0 + nbc0 + c;
          float v = (gk < K && gn < Nc) ? Bb[c] : 0.0f;
          nbdst[c * 16] = (__bf16)v;
        }
      }
      if (gk + 32 < K)
        __builtin_prefetch(Bb + (long long)32 * ldb, 0, 3);
    }
    if (Abase && k0 + 32 + ac0 < K)
      __builtin_prefetch(Abase + k0 + 32 + ac0, 0, 3);
    __syncthreads();

    v16bf af  = *(const v16bf*)(&Aswz[wm][lane][0]);
    v16bf bf0 = *(const v16bf*)(&Bswz[(wn << 1) + 0][lane][0]);
    v16bf bf1 = *(const v16bf*)(&Bswz[(wn << 1) + 1][lane][0]);
    acc0 = WMMA_BF16(af, bf0, acc0);
    acc1 = WMMA_BF16(af, bf1, acc1);
    __syncthreads();
  }

  // epilogue: C/D layout -> M = e + 8*(lane>>4), N = lane&15
  int mbase = m0 + (wm << 4) + 8 * (lane >> 4);
  int nbase = n0 + (wn << 5) + (lane & 15);
#pragma unroll
  for (int e = 0; e < 8; ++e) {
    int m = mbase + e;
    if (m >= M) continue;
#pragma unroll
    for (int t = 0; t < 2; ++t) {
      int n = nbase + t * 16;
      if (n >= Nc) continue;
      float v = alpha * (t ? acc1[e] : acc0[e]);
      if (bias) v += bias[n];
      long long idx = (long long)m * ldd + n;
      if (beta != 0.0f) v += beta * D[idx];
      if (act == 1)      v = tanhf(v);
      else if (act == 2) v = fmaxf(v, 0.0f);
      D[idx] = v;
    }
  }
}

// ---------------- attention helpers ----------------
__global__ void k_rowstats(const float* __restrict__ S, float* __restrict__ mx,
                           float* __restrict__ rden)
{
  int wave = threadIdx.x >> 5, lane = threadIdx.x & 31;
  long long row = (long long)blockIdx.x * 8 + wave;     // b*L+i
  const float* p = S + row * Ln;
  float m = -1e30f;
  for (int j = lane; j < Ln; j += 32) m = fmaxf(m, p[j]);
  for (int o = 16; o >= 1; o >>= 1) m = fmaxf(m, __shfl_xor(m, o, 32));
  float s = 0.0f;
  for (int j = lane; j < Ln; j += 32) s += __expf(p[j] - m);
  for (int o = 16; o >= 1; o >>= 1) s += __shfl_xor(s, o, 32);
  if (lane == 0) { mx[row] = m; rden[row] = 1.0f / s; }
}

__global__ void k_colsum(const float* __restrict__ S, const float* __restrict__ mx,
                         const float* __restrict__ rden, float* __restrict__ q)
{
  int b = blockIdx.y;
  int j = blockIdx.x * blockDim.x + threadIdx.x;
  const float* Sp  = S + (long long)b * Ln * Ln;
  const float* mxp = mx + b * Ln;
  const float* rdp = rden + b * Ln;
  float acc = 0.0f;
  for (int i = 0; i < Ln; ++i)
    acc += __expf(Sp[(long long)i * Ln + j] - mxp[i]) * rdp[i];
  q[b * Ln + j] = acc;
}

__global__ void k_sent(const float* __restrict__ x, const float* __restrict__ q,
                       float* __restrict__ sent)
{
  int b = blockIdx.x, d = threadIdx.x;
  if (d >= Hn) return;
  const float* xp = x + (long long)b * Ln * Hn;
  const float* qp = q + b * Ln;
  float acc = 0.0f;
  for (int m = 0; m < Ln; ++m) acc += qp[m] * xp[(long long)m * Hn + d];
  sent[b * Hn + d] = acc * (1.0f / (float)Ln);
}

// ---------------- FiLM ----------------
__global__ void k_film_a(const float* __restrict__ fs, const float* __restrict__ sk,
                         float* __restrict__ a)
{
  long long i = (long long)blockIdx.x * blockDim.x + threadIdx.x;
  if (i >= (long long)Nn * Hn) return;
  long long n = i / Hn; int d = (int)(i - n * Hn);
  float beta_  = fs[n * H2 + d];
  float gamma_ = fs[n * H2 + Hn + d];
  a[i] = fmaxf(gamma_ * sk[i] + beta_, 0.0f);
}

__global__ void k_edge(const int* __restrict__ ei, const int* __restrict__ et,
                       const float* __restrict__ fnode, const float* __restrict__ lnode,
                       float* __restrict__ sacc, float* __restrict__ cnt, int rel)
{
  int wave = threadIdx.x >> 5, lane = threadIdx.x & 31;
  long long e = (long long)blockIdx.x * 8 + wave;
  if (e >= En) return;
  if (et[e] != rel) return;
  int s = ei[e], d = ei[En + e];
  const float* fp = fnode + (long long)d * H2;
  const float* lp = lnode + (long long)s * Hn;
  float* op = sacc + (long long)d * Hn;
  for (int k = lane; k < Hn; k += 32) {
    float msg = fmaxf(fp[Hn + k] * lp[k] + fp[k], 0.0f);
    atomicAdd(&op[k], msg);
  }
  if (lane == 0) atomicAdd(&cnt[d], 1.0f);
}

__global__ void k_combine(float* __restrict__ a, const float* __restrict__ sacc,
                          const float* __restrict__ cnt)
{
  long long i = (long long)blockIdx.x * blockDim.x + threadIdx.x;
  if (i >= (long long)Nn * Hn) return;
  long long n = i / Hn;
  a[i] += sacc[i] / fmaxf(cnt[n], 1.0f);
}

// ---------------- GRU ----------------
__global__ void k_bsum(const float* __restrict__ gb, float* __restrict__ bs)
{
  int d = blockIdx.x * blockDim.x + threadIdx.x;
  if (d >= Hn) return;
  for (int g = 0; g < 3; ++g)
    bs[g * Hn + d] = gb[(3 * g + 0) * Hn + d] + gb[(3 * g + 1) * Hn + d] + gb[(3 * g + 2) * Hn + d];
}

__global__ void k_gate1(const float* __restrict__ x, float* __restrict__ rpre,
                        const float* __restrict__ szr)
{
  long long i = (long long)blockIdx.x * blockDim.x + threadIdx.x;
  if (i >= (long long)Nn * Hn) return;
  long long n = i / Hn; int d = (int)(i - n * Hn);
  int b = (int)(n >> 9);
  float rg = sigmoidf_(rpre[i] + szr[b * Hn + d]);
  rpre[i] = x[i] * rg;               // xr
}

__global__ void k_gate2(float* __restrict__ x, const float* __restrict__ zpre,
                        const float* __restrict__ hpre, const float* __restrict__ szz,
                        const float* __restrict__ szh)
{
  long long i = (long long)blockIdx.x * blockDim.x + threadIdx.x;
  if (i >= (long long)Nn * Hn) return;
  long long n = i / Hn; int d = (int)(i - n * Hn);
  int b = (int)(n >> 9);
  float z = sigmoidf_(zpre[i] + szz[b * Hn + d]);
  float h = tanhf(hpre[i] + szh[b * Hn + d]);
  x[i] = h * z + x[i] * (1.0f - z);
}

// ---------------- bf16 image of x for async LDS staging ----------------
__global__ void k_xbf16(const float* __restrict__ x, __bf16* __restrict__ xbf)
{
  long long i = (long long)blockIdx.x * blockDim.x + threadIdx.x;
  if (i >= (long long)Nn * XP) return;
  long long n = i / XP; int d = (int)(i - n * XP);
  float v = (d < Hn) ? x[n * (long long)Hn + d] : 0.0f;
  xbf[i] = (__bf16)v;
}

// =================================================================
// BiLSTM: fused K-concatenated weights, fragment-swizzled bf16 layout.
// Column permutation n' = 4*d + gate (gate order i,f,g,o).
// Wcomb [640][1200]: rows 0..299 = Wih^T, 320..619 = Whh^T, 639 = bih+bhh.
// Swizzled: idx = (((nt*20 + kc)*32 + lane)*16 + j), k = kc*32 + (lane>>4)*16 + j,
//           n' = nt*16 + (lane&15).
// =================================================================
__global__ void k_wcomb(const float* __restrict__ Wih, const float* __restrict__ Whh,
                        const float* __restrict__ bih, const float* __restrict__ bhh,
                        __bf16* __restrict__ dst)
{
  int nt = blockIdx.x / 20, kc = blockIdx.x % 20;
  int lane = threadIdx.x >> 4, j = threadIdx.x & 15;     // 512 threads
  int k = (kc << 5) + ((lane >> 4) << 4) + j;
  int n = (nt << 4) + (lane & 15);
  int dd = n >> 2, g = n & 3;
  int row = g * Hn + dd;                                  // row in [1200] x [300]
  float v = 0.0f;
  if (k < Hn)                      v = Wih[(long long)row * Hn + k];
  else if (k >= 320 && k < 320+Hn) v = Whh[(long long)row * Hn + (k - 320)];
  else if (k == 639)               v = bih[row] + bhh[row];
  dst[(((long long)(nt * 20 + kc) * 32 + lane) << 4) + j] = (__bf16)v;
}

#define LW 32
__global__ void __launch_bounds__(1024)
k_lstm(const __bf16* __restrict__ xbf, const __bf16* __restrict__ wcf,
       const __bf16* __restrict__ wcb, float* __restrict__ O)
{
  __shared__ alignas(32) __bf16 Ash[64][648];   // [x_t | pad | h | pad | 1] bf16
  __shared__ float stg[LW][16][17];             // per-wave C-tile staging

  const int dir = blockIdx.x;
  const __bf16* wc = dir ? wcb : wcf;
  const int tid = threadIdx.x;
  const int wave = tid >> 5, lane = tid & 31;

  // init cols 300..647 zero, col 639 = 1.0 (constant bias row).
  // cols 300..303 are rewritten each step by the (zero-padded) async x copy.
  for (int idx = tid; idx < 64 * 348; idx += 1024) {
    int b = idx / 348, c = 300 + (idx - (idx / 348) * 348);
    Ash[b][c] = (__bf16)((c == 639) ? 1.0f : 0.0f);
  }

  float cst[10][2];
#pragma unroll
  for (int i = 0; i < 10; ++i) { cst[i][0] = 0.0f; cst[i][1] = 0.0f; }
  v8f accs[10];

  for (int t = 0; t < Ln; ++t) {
    int l = dir ? (Ln - 1 - t) : t;
    __syncthreads();
    // stage x_t rows: 38 x 16-byte chunks per row (608 B padded bf16 rows)
    for (int idx = tid; idx < 64 * 38; idx += 1024) {
      int b = idx / 38, c = idx - (idx / 38) * 38;
      const __bf16* gsrc = xbf + ((long long)(b * Ln + l)) * XP + c * 8;
      __bf16* ldst = &Ash[b][c * 8];
#ifdef HAVE_ASYNC_LDS
      __builtin_amdgcn_global_load_async_to_lds_b128(
          (i32x4_*)(uintptr_t)gsrc, (i32x4_*)ldst, 0, 0);
#else
      *(v8bf*)ldst = *(const v8bf*)gsrc;
#endif
    }
#ifdef HAVE_ASYNC_LDS
    wait_asynccnt0();
#endif
    __syncthreads();

    // gates = [x_t | h | 1] @ Wcomb   (4 x 75 tiles of 16x16, 20 K-chunks)
    int ti = 0;
    for (int tt = wave; tt < 300; tt += LW, ++ti) {
      int mt = tt & 3, nt = tt >> 2;
      v8f acc = {0.f,0.f,0.f,0.f,0.f,0.f,0.f,0.f};
      const __bf16* arow_ = &Ash[(mt << 4) + (lane & 15)][0];
      const int kadd = (lane >> 4) << 4;
      const __bf16* wp = wc + (((long long)nt * 20 * 32 + lane) << 4);
      for (int kc = 0; kc < 20; ++kc) {
        U16bf u;
        int kb = (kc << 5) + kadd;
        u.h[0] = *(const v8bf*)(arow_ + kb);
        u.h[1] = *(const v8bf*)(arow_ + kb + 8);
        v16bf bf = *(const v16bf*)(wp + ((long long)kc * 512));
        acc = WMMA_BF16(u.v, bf, acc);
      }
      accs[ti] = acc;
    }
    __syncthreads();   // all h reads done before cell overwrites h

    // LSTM cell per owned tile (16 b x 4 d x 4 gates per tile)
    ti = 0;
    for (int tt = wave; tt < 300; tt += LW, ++ti) {
      int mt = tt & 3, nt = tt >> 2;
#pragma unroll
      for (int e = 0; e < 8; ++e)
        stg[wave][e + ((lane >> 4) << 3)][lane & 15] = accs[ti][e];
#pragma unroll
      for (int s = 0; s < 2; ++s) {
        int p  = (lane << 1) + s;      // 0..63
        int bl = p >> 2, dl = p & 3;
        float gi = stg[wave][bl][dl * 4 + 0];
        float gf = stg[wave][bl][dl * 4 + 1];
        float gg = stg[wave][bl][dl * 4 + 2];
        float go = stg[wave][bl][dl * 4 + 3];
        float cnew = sigmoidf_(gf) * cst[ti][s] + sigmoidf_(gi) * tanhf(gg);
        float h    = sigmoidf_(go) * tanhf(cnew);
        cst[ti][s] = cnew;
        int bg = (mt << 4) + bl;
        int dg = (nt << 2) + dl;
        Ash[bg][320 + dg] = (__bf16)h;
        O[((long long)(bg * Ln + l)) * H2 + dir * Hn + dg] = h;
      }
    }
  }
}

// ---------------- readout ----------------
__global__ void k_cvec(const float* __restrict__ T, const float* __restrict__ w3,
                       float* __restrict__ cvec)
{
  int wave = threadIdx.x >> 5, lane = threadIdx.x & 31;
  long long n = (long long)blockIdx.x * 8 + wave;
  if (n >= Nn) return;
  const float* tp = T + n * H2;
  float s = 0.0f;
  for (int j = lane; j < H2; j += 32) s += tp[j] * w3[j];
  for (int o = 16; o >= 1; o >>= 1) s += __shfl_xor(s, o, 32);
  if (lane == 0) cvec[n] = s;
}

__global__ void k_att(const float* __restrict__ cvec, float* __restrict__ att)
{
  __shared__ float red[32];
  int b = blockIdx.x, t = threadIdx.x;
  int wv = t >> 5, ln = t & 31;
  float v = cvec[b * Ln + t];
  float m = v;
  for (int o = 16; o >= 1; o >>= 1) m = fmaxf(m, __shfl_xor(m, o, 32));
  if (ln == 0) red[wv] = m;
  __syncthreads();
  if (t == 0) { float mm = red[0]; for (int i = 1; i < 16; ++i) mm = fmaxf(mm, red[i]); red[0] = mm; }
  __syncthreads();
  float e = __expf(v - red[0]);
  float s = e;
  for (int o = 16; o >= 1; o >>= 1) s += __shfl_xor(s, o, 32);
  if (ln == 0) red[16 + wv] = s;
  __syncthreads();
  if (t == 0) { float ss = 0.f; for (int i = 0; i < 16; ++i) ss += red[16 + i]; red[16] = ss; }
  __syncthreads();
  att[b * Ln + t] = e / red[16];
}

__global__ void k_pool(const float* __restrict__ O, const float* __restrict__ att,
                       float* __restrict__ Avec)
{
  int b = blockIdx.x, d = threadIdx.x;
  if (d >= H2) return;
  const float* ap = att + b * Ln;
  const float* op = O + (long long)b * Ln * H2 + d;
  float s = 0.0f;
  for (int l = 0; l < Ln; ++l) s += ap[l] * op[(long long)l * H2];
  Avec[b * H2 + d] = s;
}

// =================================================================
extern "C" void kernel_launch(void* const* d_in, const int* in_sizes, int n_in,
                              void* d_out, int out_size, void* d_ws, size_t ws_size,
                              hipStream_t stream)
{
  const int*   gx        = (const int*)d_in[0];
  const int*   edge_idx  = (const int*)d_in[1];
  const int*   edge_type = (const int*)d_in[2];
  const float* embed     = (const float*)d_in[3];
  const float* enc_W     = (const float*)d_in[4];
  const float* enc_b     = (const float*)d_in[5];
  const float* lins      = (const float*)d_in[6];
  const float* films_W   = (const float*)d_in[7];
  const float* films_b   = (const float*)d_in[8];
  const float* skip_W    = (const float*)d_in[9];
  const float* filmskipW = (const float*)d_in[10];
  const float* gru_W     = (const float*)d_in[11];
  const float* gru_b     = (const float*)d_in[12];
  const float* Wih_f     = (const float*)d_in[13];
  const float* Whh_f     = (const float*)d_in[14];
  const float* bih_f     = (const float*)d_in[15];
  const float* bhh_f     = (const float*)d_in[16];
  const float* Wih_b     = (const float*)d_in[17];
  const float* Whh_b     = (const float*)d_in[18];
  const float* bih_b     = (const float*)d_in[19];
  const float* bhh_b     = (const float*)d_in[20];
  const float* W3        = (const float*)d_in[21];
  const float* b3        = (const float*)d_in[22];
  const float* w3        = (const float*)d_in[23];
  const float* mlp_W     = (const float*)d_in[25];
  const float* mlp_b     = (const float*)d_in[26];

  // ---------------- workspace carve (~380 MB) ----------------
  float* ws = (float*)d_ws;
  size_t off = 0;
  auto alloc = [&](size_t n) { float* p = ws + off; off += (n + 63) & ~(size_t)63; return p; };
  float* x    = alloc((size_t)Nn * Hn);
  float* a    = alloc((size_t)Nn * Hn);   // also first half of readout T buffer
  float* sk   = alloc((size_t)Nn * Hn);   // also second half of T buffer (contiguous with a)
  float* zpre = alloc((size_t)Nn * Hn);
  float* rpre = alloc((size_t)Nn * Hn);   // also xr
  float* hpre = alloc((size_t)Nn * Hn);
  float* sacc = alloc((size_t)Nn * Hn);
  float* cnt  = alloc((size_t)Nn);
  float* t6   = alloc((size_t)Nn * H2);   // scores / fs / fnode / LSTM output O
  float* q    = alloc((size_t)Bn * Ln);
  float* mxb  = alloc((size_t)Bn * Ln);
  float* rden = alloc((size_t)Bn * Ln);
  float* sent = alloc((size_t)Bn * Hn);
  float* szz  = alloc((size_t)Bn * Hn);
  float* szr  = alloc((size_t)Bn * Hn);
  float* szh  = alloc((size_t)Bn * Hn);
  float* bsum = alloc((size_t)3 * Hn);
  float* cvec = alloc((size_t)Nn);
  float* attb = alloc((size_t)Nn);
  float* Avec = alloc((size_t)Bn * H2);
  __bf16* wcf = (__bf16*)(ws + off); off += 384000;               // 75*20*512 bf16
  __bf16* wcb = (__bf16*)(ws + off); off += 384000;
  __bf16* xbf = (__bf16*)(ws + off); off += (size_t)Nn * XP / 2;  // padded bf16 x
  float* Tbuf = a;                                                // aliases a..sk (N*600 floats)

  auto gemm = [&](const float* A, const float* Bw, const float* bias, float* D,
                  int M, int Nc, int K, int lda, int ldb, int ldd,
                  long long sA, long long sB, long long sD, int batch,
                  const int* gather, int btrans, float alpha, float beta, int act) {
    dim3 g((Nc + 63) / 64, (M + 63) / 64, batch);
    k_gemm<<<g, 256, 0, stream>>>(A, Bw, bias, D, M, Nc, K, lda, ldb, ldd,
                                  sA, sB, sD, gather, btrans, alpha, beta, act);
  };

  const int ew = 256;
  const long long NH = (long long)Nn * Hn;
  const int ewg = (int)((NH + ew - 1) / ew);

  // encode: x = tanh(embed[gx] @ enc_W + enc_b)
  gemm(embed, enc_W, enc_b, x, Nn, Hn, Hn, Hn, Hn, Hn, 0, 0, 0, 1, gx, 0, 1.0f, 0.0f, 1);

  k_bsum<<<2, 256, 0, stream>>>(gru_b, bsum);

  const float scl = 1.0f / sqrtf((float)Hn);
  for (int step = 0; step < STEPS; ++step) {
    // ---- sentence attention: S = scl * X X^T ; sent = (colsum(softmax(S))/L) @ X
    gemm(x, x, nullptr, t6, Ln, Ln, Hn, Hn, Hn, Ln,
         (long long)Ln * Hn, (long long)Ln * Hn, (long long)Ln * Ln, Bn,
         nullptr, 1, scl, 0.0f, 0);
    k_rowstats<<<Nn / 8, 256, 0, stream>>>(t6, mxb, rden);
    k_colsum<<<dim3(Ln / 128, Bn), 128, 0, stream>>>(t6, mxb, rden, q);
    k_sent<<<Bn, 320, 0, stream>>>(x, q, sent);

    // ---- FiLM skip: a = relu(gamma * (x@skip) + beta)
    gemm(x, filmskipW + (size_t)step * Hn * H2, nullptr, t6, Nn, H2, Hn, Hn, H2, H2,
         0, 0, 0, 1, nullptr, 0, 1.0f, 0.0f, 0);
    gemm(x, skip_W + (size_t)step * Hn * Hn, nullptr, sk, Nn, Hn, Hn, Hn, Hn, Hn,
         0, 0, 0, 1, nullptr, 0, 1.0f, 0.0f, 0);
    k_film_a<<<ewg, ew, 0, stream>>>(t6, sk, a);

    // ---- per-relation messages: node transforms + edge gather/scatter mean
    for (int r = 0; r < Rn; ++r) {
      gemm(x, films_W + ((size_t)(step * Rn + r)) * Hn * H2,
           films_b + (size_t)(step * Rn + r) * H2, t6, Nn, H2, Hn, Hn, H2, H2,
           0, 0, 0, 1, nullptr, 0, 1.0f, 0.0f, 0);
      gemm(x, lins + ((size_t)(step * Rn + r)) * Hn * Hn, nullptr, sk,
           Nn, Hn, Hn, Hn, Hn, Hn, 0, 0, 0, 1, nullptr, 0, 1.0f, 0.0f, 0);
      (void)hipMemsetAsync(sacc, 0, NH * sizeof(float), stream);
      (void)hipMemsetAsync(cnt, 0, (size_t)Nn * sizeof(float), stream);
      k_edge<<<En / 8, 256, 0, stream>>>(edge_idx, edge_type, t6, sk, sacc, cnt, r);
      k_combine<<<ewg, ew, 0, stream>>>(a, sacc, cnt);
    }

    // ---- GRU gates
    gemm(a, gru_W + 0 * 90000, bsum,          zpre, Nn, Hn, Hn, Hn, Hn, Hn, 0,0,0,1, nullptr,0,1.0f,0.0f,0);
    gemm(x, gru_W + 1 * 90000, nullptr,       zpre, Nn, Hn, Hn, Hn, Hn, Hn, 0,0,0,1, nullptr,0,1.0f,1.0f,0);
    gemm(a, gru_W + 3 * 90000, bsum + Hn,     rpre, Nn, Hn, Hn, Hn, Hn, Hn, 0,0,0,1, nullptr,0,1.0f,0.0f,0);
    gemm(x, gru_W + 4 * 90000, nullptr,       rpre, Nn, Hn, Hn, Hn, Hn, Hn, 0,0,0,1, nullptr,0,1.0f,1.0f,0);
    gemm(sent, gru_W + 2 * 90000, nullptr, szz, Bn, Hn, Hn, Hn, Hn, Hn, 0,0,0,1, nullptr,0,1.0f,0.0f,0);
    gemm(sent, gru_W + 5 * 90000, nullptr, szr, Bn, Hn, Hn, Hn, Hn, Hn, 0,0,0,1, nullptr,0,1.0f,0.0f,0);
    gemm(sent, gru_W + 8 * 90000, nullptr, szh, Bn, Hn, Hn, Hn, Hn, Hn, 0,0,0,1, nullptr,0,1.0f,0.0f,0);
    k_gate1<<<ewg, ew, 0, stream>>>(x, rpre, szr);   // rpre := x * sigmoid(rpre + szr)
    gemm(a,    gru_W + 6 * 90000, bsum + 2*Hn, hpre, Nn, Hn, Hn, Hn, Hn, Hn, 0,0,0,1, nullptr,0,1.0f,0.0f,0);
    gemm(rpre, gru_W + 7 * 90000, nullptr,     hpre, Nn, Hn, Hn, Hn, Hn, Hn, 0,0,0,1, nullptr,0,1.0f,1.0f,0);
    k_gate2<<<ewg, ew, 0, stream>>>(x, zpre, hpre, szz, szh);
  }

  // ---- BiLSTM (persistent fused kernel, one WG per direction)
  k_wcomb<<<75 * 20, 512, 0, stream>>>(Wih_f, Whh_f, bih_f, bhh_f, wcf);
  k_wcomb<<<75 * 20, 512, 0, stream>>>(Wih_b, Whh_b, bih_b, bhh_b, wcb);
  {
    long long tot = (long long)Nn * XP;
    k_xbf16<<<(int)((tot + 255) / 256), 256, 0, stream>>>(x, xbf);
  }
  k_lstm<<<2, 1024, 0, stream>>>(xbf, wcf, wcb, t6);   // t6 = O [B,L,600]

  // ---- readout
  gemm(t6, W3, b3, Tbuf, Nn, H2, H2, H2, H2, H2, 0, 0, 0, 1, nullptr, 0, 1.0f, 0.0f, 1);
  k_cvec<<<Nn / 8, 256, 0, stream>>>(Tbuf, w3, cvec);     // w3b cancels in softmax
  k_att<<<Bn, 512, 0, stream>>>(cvec, attb);
  k_pool<<<Bn, 640, 0, stream>>>(t6, attb, Avec);
  gemm(Avec, mlp_W, mlp_b, (float*)d_out, Bn, Cn, H2, H2, Cn, Cn,
       0, 0, 0, 1, nullptr, 0, 1.0f, 0.0f, 0);

  (void)in_sizes; (void)n_in; (void)out_size; (void)ws_size;
}
